// QuestionEmbedding_30760555774661
// MI455X (gfx1250) — matrix operand order repaired and checked by reading, not compiled, not run on hardware
//
#include <hip/hip_runtime.h>
#include <hip/hip_bf16.h>

// ---------------------------------------------------------------------------
// QuestionEmbedding + LSTM (last hidden) for MI455X / gfx1250, bf16 WMMA path
//   VOCAB=20000, EMB=512, HID=1024, B=128, W=26
// Fused per-timestep kernel: gates GEMM (v_wmma_f32_16x16x32_bf16, f32 acc)
// with software-pipelined global loads + in-register gate nonlinearity.
// ---------------------------------------------------------------------------

#define VOCAB 20000
#define EMB   512
#define HID   1024
#define BATCH 128
#define SEQ   26
#define KTOT  (EMB + HID)        // 1536
#define GATES (4 * HID)          // 4096

typedef __bf16 v16bf __attribute__((ext_vector_type(16)));
typedef __bf16 bfx8  __attribute__((ext_vector_type(8)));
typedef float  v8f   __attribute__((ext_vector_type(8)));

static __device__ __forceinline__ float sigmoidf_(float x) {
    return 1.0f / (1.0f + __expf(-x));
}

// A fragment (16x32 bf16, M x K): lane holds 16 contiguous K {koff..koff+15}
static __device__ __forceinline__ v16bf load_a(const __bf16* ap, int akoff) {
    bfx8 lo = *(const bfx8*)(ap + akoff);
    bfx8 hi = *(const bfx8*)(ap + akoff + 8);
    union { v16bf v; bfx8 h[2]; } u;
    u.h[0] = lo; u.h[1] = hi;
    return u.v;
}

// B fragment (32x16 bf16, K x N): lane holds 16 contiguous K of weight row n
static __device__ __forceinline__ v16bf load_b(const __bf16* bp) {
    return *(const v16bf*)bp;
}

// --- setup: fuse W_ih | W_hh into bf16 Wz[4096][1536] ----------------------
__global__ __launch_bounds__(256) void prep_weights(
    const float* __restrict__ Wih,   // [4096][512]
    const float* __restrict__ Whh,   // [4096][1024]
    __bf16* __restrict__ Wz)         // [4096][1536]
{
    int gid = blockIdx.x * 256 + threadIdx.x;          // < 4096*1536
    int n = gid / KTOT;
    int k = gid - n * KTOT;
    float v = (k < EMB) ? Wih[n * EMB + k] : Whh[n * HID + (k - EMB)];
    Wz[gid] = (__bf16)v;
}

__global__ __launch_bounds__(256) void prep_bias(
    const float* __restrict__ bih, const float* __restrict__ bhh,
    float* __restrict__ bcomb)
{
    int gid = blockIdx.x * 256 + threadIdx.x;          // < 4096
    bcomb[gid] = bih[gid] + bhh[gid];
}

// --- embedding gather + tanh -> bf16, layout [SEQ][BATCH][EMB] -------------
__global__ __launch_bounds__(256) void embed_kernel(
    const int*   __restrict__ ques,     // [BATCH][SEQ]
    const float* __restrict__ Wlook,    // [EMB][VOCAB]
    __bf16*      __restrict__ emb)      // [SEQ][BATCH][EMB]
{
    int gid = blockIdx.x * 256 + threadIdx.x;          // < SEQ*BATCH*EMB
    int e = gid & (EMB - 1);
    int b = (gid >> 9) & (BATCH - 1);
    int t = gid >> 16;                                 // / (512*128)
    int q = ques[b * SEQ + t];
    float v = 0.0f;
    if (q > 0) v = tanhf(Wlook[(size_t)e * VOCAB + (q - 1)]);
    emb[gid] = (__bf16)v;
}

// --- zero h0 (bf16) and c0 (f32) -------------------------------------------
__global__ __launch_bounds__(256) void init_state(
    __bf16* __restrict__ h0, float* __restrict__ c0)
{
    int gid = blockIdx.x * 256 + threadIdx.x;          // < BATCH*HID
    h0[gid] = (__bf16)0.0f;
    c0[gid] = 0.0f;
}

// --- fused LSTM step --------------------------------------------------------
// grid = 64 blocks (one 16-col slice of HID each), 256 threads = 8 waves.
// wave w owns batch rows [16w,16w+16) and the i,f,g,o gate tiles of its slice.
// K loop is split emb-phase / h-phase and software-pipelined one stage deep.
__global__ __launch_bounds__(256) void lstm_step(
    const __bf16* __restrict__ emb_t,   // [BATCH][EMB]
    const __bf16* __restrict__ h_in,    // [BATCH][HID]
    const __bf16* __restrict__ Wz,      // [GATES][KTOT]
    const float*  __restrict__ bcomb,   // [GATES]
    float*        __restrict__ c,       // [BATCH][HID]
    __bf16*       __restrict__ h_out,   // [BATCH][HID]
    float*        __restrict__ out,     // [BATCH][HID]
    int           write_out)
{
    const int lane  = threadIdx.x & 31;
    const int wv    = threadIdx.x >> 5;        // 0..7 : batch row tile
    const int n0    = blockIdx.x;              // 0..63 : hidden col tile
    const int nloc  = lane & 15;
    const int akoff = (lane < 16) ? 0 : 16;    // A lane-half K offset
    const int bkoff = (lane < 16) ? 0 : 16;    // B lane-half K offset
    const int arow  = wv * 16 + nloc;          // batch row held by this lane

    const __bf16* aemb = emb_t + (size_t)arow * EMB;   // A phase-1 base
    const __bf16* ah   = h_in  + (size_t)arow * HID;   // A phase-2 base
    const __bf16* br0 = Wz + (size_t)(0 * HID + n0 * 16 + nloc) * KTOT + bkoff;
    const __bf16* br1 = Wz + (size_t)(1 * HID + n0 * 16 + nloc) * KTOT + bkoff;
    const __bf16* br2 = Wz + (size_t)(2 * HID + n0 * 16 + nloc) * KTOT + bkoff;
    const __bf16* br3 = Wz + (size_t)(3 * HID + n0 * 16 + nloc) * KTOT + bkoff;

    v8f acc0 = {}, acc1 = {}, acc2 = {}, acc3 = {};

    // ---- prologue: k = 0 fragments ----
    v16bf a_c  = load_a(aemb, akoff);
    v16bf b0_c = load_b(br0);
    v16bf b1_c = load_b(br1);
    v16bf b2_c = load_b(br2);
    v16bf b3_c = load_b(br3);

    int k = 0;
    // ---- phase 1: K in [0, 512) from emb_t ----
    for (; k < EMB; k += 32) {
        const int kn = k + 32;
        const __bf16* apn = (kn < EMB) ? (aemb + kn) : ah;  // crosses into h at kn==512
        v16bf a_n  = load_a(apn, akoff);
        v16bf b0_n = load_b(br0 + kn);
        v16bf b1_n = load_b(br1 + kn);
        v16bf b2_n = load_b(br2 + kn);
        v16bf b3_n = load_b(br3 + kn);
        acc0 = __builtin_amdgcn_wmma_f32_16x16x32_bf16(false, a_c, false, b0_c,
                                                       (short)0, acc0, false, false);
        acc1 = __builtin_amdgcn_wmma_f32_16x16x32_bf16(false, a_c, false, b1_c,
                                                       (short)0, acc1, false, false);
        acc2 = __builtin_amdgcn_wmma_f32_16x16x32_bf16(false, a_c, false, b2_c,
                                                       (short)0, acc2, false, false);
        acc3 = __builtin_amdgcn_wmma_f32_16x16x32_bf16(false, a_c, false, b3_c,
                                                       (short)0, acc3, false, false);
        a_c = a_n; b0_c = b0_n; b1_c = b1_n; b2_c = b2_n; b3_c = b3_n;
    }
    // ---- phase 2: K in [512, 1536) from h_in ----
    for (; k < KTOT; k += 32) {
        const int kn = k + 32;
        v16bf a_n = a_c, b0_n = b0_c, b1_n = b1_c, b2_n = b2_c, b3_n = b3_c;
        if (kn < KTOT) {
            a_n  = load_a(ah + (kn - EMB), akoff);
            b0_n = load_b(br0 + kn);
            b1_n = load_b(br1 + kn);
            b2_n = load_b(br2 + kn);
            b3_n = load_b(br3 + kn);
        }
        acc0 = __builtin_amdgcn_wmma_f32_16x16x32_bf16(false, a_c, false, b0_c,
                                                       (short)0, acc0, false, false);
        acc1 = __builtin_amdgcn_wmma_f32_16x16x32_bf16(false, a_c, false, b1_c,
                                                       (short)0, acc1, false, false);
        acc2 = __builtin_amdgcn_wmma_f32_16x16x32_bf16(false, a_c, false, b2_c,
                                                       (short)0, acc2, false, false);
        acc3 = __builtin_amdgcn_wmma_f32_16x16x32_bf16(false, a_c, false, b3_c,
                                                       (short)0, acc3, false, false);
        a_c = a_n; b0_c = b0_n; b1_c = b1_n; b2_c = b2_n; b3_c = b3_n;
    }

    // ---- epilogue: bias + LSTM pointwise, fully in registers ----
    const int col   = n0 * 16 + nloc;          // hidden index j in [0,1024)
    const int mhalf = (lane < 16) ? 0 : 8;     // C/D layout: VGPR r <-> M = r + mhalf
    const float bi = bcomb[0 * HID + col];
    const float bf = bcomb[1 * HID + col];
    const float bg = bcomb[2 * HID + col];
    const float bo = bcomb[3 * HID + col];

#pragma unroll
    for (int r = 0; r < 8; ++r) {
        const int row = wv * 16 + r + mhalf;   // batch row
        const size_t idx = (size_t)row * HID + col;
        float iv = acc0[r] + bi;
        float fv = acc1[r] + bf;
        float gv = acc2[r] + bg;
        float ov = acc3[r] + bo;
        float cn = sigmoidf_(fv) * c[idx] + sigmoidf_(iv) * tanhf(gv);
        float hn = sigmoidf_(ov) * tanhf(cn);
        c[idx]     = cn;
        h_out[idx] = (__bf16)hn;
        if (write_out) out[idx] = hn;
    }
}

// ---------------------------------------------------------------------------
extern "C" void kernel_launch(void* const* d_in, const int* in_sizes, int n_in,
                              void* d_out, int out_size, void* d_ws, size_t ws_size,
                              hipStream_t stream) {
    (void)in_sizes; (void)n_in; (void)out_size; (void)ws_size;

    const int*   ques  = (const int*)d_in[0];     // [128][26]
    const float* Wlook = (const float*)d_in[1];   // [512][20000]
    const float* Wih   = (const float*)d_in[2];   // [4096][512]
    const float* Whh   = (const float*)d_in[3];   // [4096][1024]
    const float* bih   = (const float*)d_in[4];   // [4096]
    const float* bhh   = (const float*)d_in[5];   // [4096]
    float*       out   = (float*)d_out;           // [128][1024]

    // workspace layout (256B aligned)
    char* ws = (char*)d_ws;
    size_t off = 0;
    __bf16* Wz    = (__bf16*)(ws + off); off += (size_t)GATES * KTOT * 2;        // 12.6 MB
    float*  bcomb = (float*) (ws + off); off += (size_t)GATES * 4;               // 16 KB
    off = (off + 255) & ~(size_t)255;
    __bf16* emb   = (__bf16*)(ws + off); off += (size_t)SEQ * BATCH * EMB * 2;   // 3.4 MB
    off = (off + 255) & ~(size_t)255;
    __bf16* hA    = (__bf16*)(ws + off); off += (size_t)BATCH * HID * 2;
    off = (off + 255) & ~(size_t)255;
    __bf16* hB    = (__bf16*)(ws + off); off += (size_t)BATCH * HID * 2;
    off = (off + 255) & ~(size_t)255;
    float*  cst   = (float*) (ws + off); off += (size_t)BATCH * HID * 4;

    prep_weights<<<(GATES * KTOT) / 256, 256, 0, stream>>>(Wih, Whh, Wz);
    prep_bias<<<GATES / 256, 256, 0, stream>>>(bih, bhh, bcomb);
    embed_kernel<<<(SEQ * BATCH * EMB) / 256, 256, 0, stream>>>(ques, Wlook, emb);
    init_state<<<(BATCH * HID) / 256, 256, 0, stream>>>(hA, cst);

    __bf16* hin  = hA;
    __bf16* hout = hB;
    for (int t = 0; t < SEQ; ++t) {
        const __bf16* emb_t = emb + (size_t)t * BATCH * EMB;
        lstm_step<<<HID / 16, 256, 0, stream>>>(emb_t, hin, Wz, bcomb, cst, hout, out,
                                                (t == SEQ - 1) ? 1 : 0);
        __bf16* tmp = hin; hin = hout; hout = tmp;
    }
}
